// AttentionHead_28389733827022
// MI455X (gfx1250) — compile-verified, matrix operand order checked by reading, not run on hardware
//
#include <hip/hip_runtime.h>
#include <hip/hip_bf16.h>
#include <math.h>

// Problem constants (B, S, E, D) = (8, 2048, 768, 64)
static constexpr int Bn = 8;
static constexpr int Sn = 2048;
static constexpr int En = 768;
static constexpr int Dn = 64;

typedef __attribute__((ext_vector_type(16))) _Float16 v16h;
typedef __attribute__((ext_vector_type(8)))  _Float16 v8h;
typedef __attribute__((ext_vector_type(8)))  float    v8f;

// ---------------------------------------------------------------------------
// Fragment loaders (CDNA5 WMMA 16x16x32 f16 layout, ISA 7.12.2):
//   lane L: index = L%16 (A: row M, B: col N), h = L/16
//   v16h elements 0..7  <- K = h*8 .. h*8+7
//   v16h elements 8..15 <- K = h*8+16 .. h*8+23
// Works for both global and LDS source pointers (flat addressing).
// ---------------------------------------------------------------------------
__device__ __forceinline__ v16h load_frag_h(const _Float16* p, int h) {
    const int kb = h * 8;
    v8h lo = *(const v8h*)(p + kb);
    v8h hi = *(const v8h*)(p + kb + 16);
    return __builtin_shufflevector(lo, hi, 0,1,2,3,4,5,6,7,8,9,10,11,12,13,14,15);
}

__device__ __forceinline__ v16h load_frag_f32(const float* __restrict__ p, int h) {
    const int kb = h * 8;
    v16h a;
#pragma unroll
    for (int e = 0; e < 8; ++e) {
        a[e]     = (_Float16)p[kb + e];
        a[e + 8] = (_Float16)p[kb + 16 + e];
    }
    return a;
}

// ---------------------------------------------------------------------------
// K0: transpose + f32->f16 convert weights: W[E,D] -> Wt[D,E]
// ---------------------------------------------------------------------------
__global__ void prep_weights_kernel(const float* __restrict__ Wq,
                                    const float* __restrict__ Wk,
                                    _Float16* __restrict__ Wqt,
                                    _Float16* __restrict__ Wkt) {
    const int idx = blockIdx.x * blockDim.x + threadIdx.x;   // over E*D
    const int d = idx % Dn;
    const int e = idx / Dn;
    if (blockIdx.y == 0) Wqt[(size_t)d * En + e] = (_Float16)Wq[idx];
    else                 Wkt[(size_t)d * En + e] = (_Float16)Wk[idx];
}

// ---------------------------------------------------------------------------
// K1: projections. C[M=B*S, N=64] = X[M,E] @ W[E,64] + bias.
// blockIdx.y: 0 -> q (also writes transposed copy qT used as "v"), 1 -> k.
// 256 threads = 8 waves, 128 rows per workgroup, full N=64.
// Weight slab staged through LDS in 64x256 chunks (32 KB), shared by 8 waves.
// ---------------------------------------------------------------------------
__global__ void __launch_bounds__(256)
proj_kernel(const float* __restrict__ X,
            const _Float16* __restrict__ Wqt, const float* __restrict__ bq,
            const _Float16* __restrict__ Wkt, const float* __restrict__ bk,
            _Float16* __restrict__ qh, _Float16* __restrict__ kh,
            _Float16* __restrict__ qth) {
    __shared__ __align__(16) _Float16 wtile[64 * 256];   // 32 KB

    const int tid  = threadIdx.x;
    const int lane = tid & 31, wave = tid >> 5;
    const int h = lane >> 4, ln = lane & 15;
    const bool isQ = (blockIdx.y == 0);
    const _Float16* Wt   = isQ ? Wqt : Wkt;
    const float*    bias = isQ ? bq  : bk;
    _Float16*       outh = isQ ? qh  : kh;

    const int m0 = blockIdx.x * 128 + wave * 16;
    const float* xrow = X + (size_t)(m0 + ln) * En;

    v8f acc[4] = {};
    for (int kc = 0; kc < En; kc += 256) {
        // Stage Wt[0..63][kc..kc+255] -> LDS (rows strided En in global)
#pragma unroll
        for (int i = tid; i < (64 * 256) / 8; i += 256) {
            const int row = i >> 5;            // 32 v8h chunks per row
            const int c8  = (i & 31) * 8;
            *(v8h*)(wtile + row * 256 + c8) =
                *(const v8h*)(Wt + (size_t)row * En + kc + c8);
        }
        __syncthreads();
#pragma unroll
        for (int kl = 0; kl < 256; kl += 32) {
            v16h a = load_frag_f32(xrow + kc + kl, h);
#pragma unroll
            for (int nt = 0; nt < 4; ++nt) {
                v16h bfrag = load_frag_h(wtile + (nt * 16 + ln) * 256 + kl, h);
                acc[nt] = __builtin_amdgcn_wmma_f32_16x16x32_f16(
                    false, a, false, bfrag, (short)0, acc[nt], false, false);
            }
        }
        __syncthreads();
    }

    // Store: tile never straddles a batch boundary (Sn % 128 == 0)
    const int bi  = m0 >> 11;          // m0 / Sn
    const int si0 = m0 & (Sn - 1);     // m0 % Sn
#pragma unroll
    for (int nt = 0; nt < 4; ++nt) {
        const int n = nt * 16 + ln;
        const float bv = bias[n];
        _Float16 vals[8];
#pragma unroll
        for (int r = 0; r < 8; ++r) {
            vals[r] = (_Float16)(acc[nt][r] + bv);
            outh[(size_t)(m0 + r + h * 8) * Dn + n] = vals[r];
        }
        if (isQ) {
            _Float16* qtrow = qth + ((size_t)bi * Dn + n) * Sn + si0 + h * 8;
#pragma unroll
            for (int r = 0; r < 8; ++r) qtrow[r] = vals[r];   // [B, D, S]
        }
    }
}

// ---------------------------------------------------------------------------
// K2: scores[b,q,kcol] = scale * sum_d q[b,q,d]*k[b,kcol,d]   (fp32 out)
// Per workgroup: 128(q) x 128(kcol) tile, K=64 -> 2 WMMA steps per tile.
// k-tile (128x64 f16, 16 KB, contiguous) staged once in LDS for all 8 waves.
// ---------------------------------------------------------------------------
__global__ void __launch_bounds__(256)
scores_kernel(const _Float16* __restrict__ qh,
              const _Float16* __restrict__ kh,
              float* __restrict__ scores, float scale) {
    __shared__ __align__(16) _Float16 ktile[128 * 64];   // 16 KB

    const int tid  = threadIdx.x;
    const int lane = tid & 31, wave = tid >> 5;
    const int h = lane >> 4, ln = lane & 15;
    const int b  = blockIdx.z;
    const int m0 = blockIdx.y * 128 + wave * 16;
    const int n0 = blockIdx.x * 128;

    const _Float16* qb = qh + (size_t)b * Sn * Dn;
    const _Float16* ksrc = kh + (size_t)b * Sn * Dn + (size_t)n0 * Dn;

    // Contiguous 16 KB cooperative copy
#pragma unroll
    for (int i = tid; i < (128 * 64) / 8; i += 256)
        ((v8h*)ktile)[i] = ((const v8h*)ksrc)[i];
    __syncthreads();

    const _Float16* arow = qb + (size_t)(m0 + ln) * Dn;
    v16h a0 = load_frag_h(arow,      h);   // K(d) = 0..31
    v16h a1 = load_frag_h(arow + 32, h);   // K(d) = 32..63

    v8f acc[8] = {};
#pragma unroll
    for (int nt = 0; nt < 8; ++nt) {
        const _Float16* brow = ktile + (nt * 16 + ln) * Dn;
        v16h b0 = load_frag_h(brow,      h);
        v16h b1 = load_frag_h(brow + 32, h);
        acc[nt] = __builtin_amdgcn_wmma_f32_16x16x32_f16(
            false, a0, false, b0, (short)0, acc[nt], false, false);
        acc[nt] = __builtin_amdgcn_wmma_f32_16x16x32_f16(
            false, a1, false, b1, (short)0, acc[nt], false, false);
    }
    float* sb = scores + (size_t)b * Sn * Sn;
#pragma unroll
    for (int nt = 0; nt < 8; ++nt) {
        const int n = n0 + nt * 16 + ln;
#pragma unroll
        for (int r = 0; r < 8; ++r) {
            const int m = m0 + r + h * 8;
            sb[(size_t)m * Sn + n] = acc[nt][r] * scale;
        }
    }
}

// ---------------------------------------------------------------------------
// K3: column softmax stats over the q axis (reference: softmax(axis=1)).
// One thread per (b, kcol); online max / sum-exp. Coalesced across kcol.
// ---------------------------------------------------------------------------
__global__ void col_stats_kernel(const float* __restrict__ scores,
                                 float* __restrict__ mcol,
                                 float* __restrict__ rz) {
    const int b = blockIdx.y;
    const int k = blockIdx.x * blockDim.x + threadIdx.x;
    const float* col = scores + (size_t)b * Sn * Sn + k;
    float m = -3.4e38f, z = 0.0f;
    for (int q = 0; q < Sn; ++q) {
        const float s = col[(size_t)q * Sn];
        if (s > m) { z = z * __expf(m - s) + 1.0f; m = s; }
        else       { z += __expf(s - m); }
    }
    mcol[(size_t)b * Sn + k] = m;
    rz  [(size_t)b * Sn + k] = 1.0f / z;
}

// ---------------------------------------------------------------------------
// K4: out[b,q,d] = sum_k softmax_col(scores)[q,k] * v[k,d]   (v = q, f16 [B,D,S])
// A fragment (P = exp(s-m)*rZ) built on the fly in fp32, converted to f16.
// Per workgroup: 128(q) x 64(d); K loop over 2048. v staged through LDS in
// 64x256 chunks (32 KB), shared by all 8 waves.
// ---------------------------------------------------------------------------
__global__ void __launch_bounds__(256)
out_kernel(const float* __restrict__ scores,
           const float* __restrict__ mcol,
           const float* __restrict__ rz,
           const _Float16* __restrict__ vth,
           float* __restrict__ out) {
    __shared__ __align__(16) _Float16 vtile[64 * 256];   // 32 KB

    const int tid  = threadIdx.x;
    const int lane = tid & 31, wave = tid >> 5;
    const int h = lane >> 4, ln = lane & 15;
    const int b  = blockIdx.y;
    const int m0 = blockIdx.x * 128 + wave * 16;

    const float* srow = scores + ((size_t)b * Sn + (m0 + ln)) * Sn;
    const float* mrow = mcol + (size_t)b * Sn;
    const float* zrow = rz   + (size_t)b * Sn;
    const _Float16* vb = vth + (size_t)b * Dn * Sn;

    v8f acc[4] = {};
    for (int kc = 0; kc < Sn; kc += 256) {
        // Stage v[0..63][kc..kc+255] -> LDS (rows strided Sn in global)
#pragma unroll
        for (int i = tid; i < (64 * 256) / 8; i += 256) {
            const int row = i >> 5;
            const int c8  = (i & 31) * 8;
            *(v8h*)(vtile + row * 256 + c8) =
                *(const v8h*)(vb + (size_t)row * Sn + kc + c8);
        }
        __syncthreads();
#pragma unroll
        for (int kl = 0; kl < 256; kl += 32) {
            const int kk = kc + kl;
            if (kk + 32 < Sn) __builtin_prefetch(srow + kk + 32, 0, 1);  // global_prefetch_b8
            const int kb = kk + h * 8;
            v16h a;
#pragma unroll
            for (int e = 0; e < 8; ++e) {
                const int k0 = kb + e, k1 = kb + 16 + e;
                a[e]     = (_Float16)(__expf(srow[k0] - mrow[k0]) * zrow[k0]);
                a[e + 8] = (_Float16)(__expf(srow[k1] - mrow[k1]) * zrow[k1]);
            }
#pragma unroll
            for (int nt = 0; nt < 4; ++nt) {
                v16h bfrag = load_frag_h(vtile + (nt * 16 + ln) * 256 + kl, h);
                acc[nt] = __builtin_amdgcn_wmma_f32_16x16x32_f16(
                    false, a, false, bfrag, (short)0, acc[nt], false, false);
            }
        }
        __syncthreads();
    }
#pragma unroll
    for (int nt = 0; nt < 4; ++nt) {
        const int n = nt * 16 + ln;
#pragma unroll
        for (int r = 0; r < 8; ++r) {
            const int m = m0 + r + h * 8;
            out[((size_t)b * Sn + m) * Dn + n] = acc[nt][r];
        }
    }
}

// ---------------------------------------------------------------------------
// Launch
// ---------------------------------------------------------------------------
extern "C" void kernel_launch(void* const* d_in, const int* in_sizes, int n_in,
                              void* d_out, int out_size, void* d_ws, size_t ws_size,
                              hipStream_t stream) {
    (void)in_sizes; (void)n_in; (void)out_size; (void)ws_size;

    const float* X  = (const float*)d_in[0];   // input_ids [B,S,E]
    const float* Wq = (const float*)d_in[1];   // [E,D]
    const float* bq = (const float*)d_in[2];   // [D]
    const float* Wk = (const float*)d_in[3];   // [E,D]
    const float* bk = (const float*)d_in[4];   // [D]
    // d_in[5]/d_in[6] (Wv, bv) intentionally unused: reference uses v = q_net(x).
    float* out = (float*)d_out;                // [B,S,D] fp32

    // Workspace layout (bytes), 256B-aligned chunks. Total ~141 MB.
    char* ws = (char*)d_ws;
    const size_t szQK   = (size_t)Bn * Sn * Dn * sizeof(_Float16);   // 2 MB each
    const size_t szWt   = ((size_t)En * Dn * sizeof(_Float16) + 255) & ~(size_t)255;
    const size_t szStat = ((size_t)Bn * Sn * sizeof(float) + 255) & ~(size_t)255;
    size_t off = 0;
    _Float16* qh  = (_Float16*)(ws + off); off += szQK;
    _Float16* kh  = (_Float16*)(ws + off); off += szQK;
    _Float16* qth = (_Float16*)(ws + off); off += szQK;   // [B,D,S], used as v
    _Float16* Wqt = (_Float16*)(ws + off); off += szWt;
    _Float16* Wkt = (_Float16*)(ws + off); off += szWt;
    float* mcol   = (float*)(ws + off);    off += szStat;
    float* rz     = (float*)(ws + off);    off += szStat;
    float* scores = (float*)(ws + off);    // [B,S,S] fp32, 134 MB (fits in 192MB L2)

    const float scale = 1.0f / sqrtf((float)Sn);   // 1/sqrt(seq_len), per reference

    // K0: weight transpose/convert
    prep_weights_kernel<<<dim3((En * Dn) / 256, 2), 256, 0, stream>>>(Wq, Wk, Wqt, Wkt);
    // K1: q/k projections (+ transposed q for use as v)
    proj_kernel<<<dim3((Bn * Sn) / 128, 2), 256, 0, stream>>>(
        X, Wqt, bq, Wkt, bk, qh, kh, qth);
    // K2: scaled scores, fp32
    scores_kernel<<<dim3(Sn / 128, Sn / 128, Bn), 256, 0, stream>>>(qh, kh, scores, scale);
    // K3: column (query-axis) softmax stats
    col_stats_kernel<<<dim3(Sn / 256, Bn), 256, 0, stream>>>(scores, mcol, rz);
    // K4: out = softmax_col(scores) @ v
    out_kernel<<<dim3(Sn / 128, Bn), 256, 0, stream>>>(scores, mcol, rz, qth, out);
}
